// MultisourceConvAttention_63634235457682
// MI455X (gfx1250) — compile-verified, matrix operand order checked by reading, not run on hardware
//
#include <hip/hip_runtime.h>
#include <hip/hip_bf16.h>
#include <math.h>

// ---------------------------------------------------------------------------
// Problem constants (from the reference)
// ---------------------------------------------------------------------------
#define BB    4          // batch
#define NHEAD 8
#define CCH   64
#define ADIM  256        // per-head embedding dim
#define DHEAD 128        // per-head patch dim (PH*PW*C/NH)
#define DP    1024       // full patch feature dim (PH*PW*C)
#define NK0   1024       // key0 patches (32x32)
#define NK1   576        // key1 patches (24x24)
#define NKT   1600
#define NQ0   1024       // query0 patches (32x32)
#define NQ1   256        // query1 patches (16x16)
#define NQT   1280
#define HID   256

typedef __bf16 bf16_t;
typedef __attribute__((ext_vector_type(16))) __bf16 v16bf;
typedef __attribute__((ext_vector_type(8)))  __bf16 v8bf;
typedef __attribute__((ext_vector_type(8)))  float  v8f;

// ---------------------------------------------------------------------------
// WMMA helpers (CDNA5 v_wmma_f32_16x16x32_bf16, wave32)
//
// Fragment layouts follow CDNA5 ISA:
//   A (16x32, MxK): lane = half*16+m holds row m, 16 bf16 elems covering
//     K = half*8 + [0..7]  and  K = 16 + half*8 + [0..7]   (two 16B runs)
//   B (32x16, KxN): stored as Bt[N][K] row-major -> identical loader (N<->M).
//   C/D (16x16 f32): lane n = lane&15, VGPR r -> M = (lane>>4)*8 + r.
// ---------------------------------------------------------------------------
__device__ inline v8f wmma_bf16(v16bf a, v16bf b, v8f c) {
    return __builtin_amdgcn_wmma_f32_16x16x32_bf16(false, a, false, b,
                                                   (short)0, c, false, false);
}

// Load a 16x32 fragment from a row-major bf16 matrix (rows = M or N, cols = K).
// Requires: base 16B aligned, ld multiple of 8, k0 multiple of 32.
__device__ inline v16bf load_mk_frag(const bf16_t* base, int ld, int r0, int k0) {
    const int lane = threadIdx.x & 31;
    const int m    = lane & 15;
    const int h    = lane >> 4;
    const bf16_t* p = base + (size_t)(r0 + m) * (size_t)ld + k0 + h * 8;
    v8bf lo = *(const v8bf*)(p);
    v8bf hi = *(const v8bf*)(p + 16);
    v16bf out;
#pragma unroll
    for (int i = 0; i < 8; ++i) { out[i] = lo[i]; out[i + 8] = hi[i]; }
    return out;
}

// Load a B fragment from a [K][N] row-major *float* matrix (converted to bf16
// on the fly).  Used for conv1 where activations are NCHW (pixels along N).
__device__ inline v16bf load_kn_frag_f32(const float* Bkn, size_t ldb, int k0, int n0) {
    const int lane = threadIdx.x & 31;
    const int n    = lane & 15;
    const int h    = lane >> 4;
    v16bf out;
#pragma unroll
    for (int i = 0; i < 16; ++i) {
        const int k = (i < 8) ? (h * 8 + i) : (16 + h * 8 + (i - 8));
        out[i] = (bf16_t)Bkn[(size_t)(k0 + k) * ldb + n0 + n];
    }
    return out;
}

// ---------------------------------------------------------------------------
// 32x32 wave tile (used where K is short: embeddings K=128, conv1 K=128)
// ---------------------------------------------------------------------------
struct Acc32 { v8f c[2][2]; };

__device__ inline Acc32 wave_gemm32x32(const bf16_t* A, int lda,
                                       const bf16_t* Bt, int ldb,
                                       int m0, int n0, int klen) {
    Acc32 r;
#pragma unroll
    for (int i = 0; i < 2; ++i)
#pragma unroll
        for (int j = 0; j < 2; ++j)
            r.c[i][j] = (v8f){0.f,0.f,0.f,0.f,0.f,0.f,0.f,0.f};
    for (int k = 0; k < klen; k += 32) {
        v16bf a0 = load_mk_frag(A,  lda, m0,      k);
        v16bf a1 = load_mk_frag(A,  lda, m0 + 16, k);
        v16bf b0 = load_mk_frag(Bt, ldb, n0,      k);
        v16bf b1 = load_mk_frag(Bt, ldb, n0 + 16, k);
        r.c[0][0] = wmma_bf16(a0, b0, r.c[0][0]);
        r.c[0][1] = wmma_bf16(a0, b1, r.c[0][1]);
        r.c[1][0] = wmma_bf16(a1, b0, r.c[1][0]);
        r.c[1][1] = wmma_bf16(a1, b1, r.c[1][1]);
    }
    return r;
}

// ---------------------------------------------------------------------------
// 64x64 wave tile for the big-K GEMMs (QK^T, P@V): 16 accumulators,
// 8 fragment loads per 16 WMMAs -> 1:1 b128-load:wmma issue ratio.
// ---------------------------------------------------------------------------
struct Acc64 { v8f c[4][4]; };

__device__ inline Acc64 wave_gemm64x64(const bf16_t* A, int lda,
                                       const bf16_t* Bt, int ldb,
                                       int m0, int n0, int klen) {
    Acc64 r;
#pragma unroll
    for (int i = 0; i < 4; ++i)
#pragma unroll
        for (int j = 0; j < 4; ++j)
            r.c[i][j] = (v8f){0.f,0.f,0.f,0.f,0.f,0.f,0.f,0.f};
    for (int k = 0; k < klen; k += 32) {
        if (k + 32 < klen) {   // prefetch next K-slice (global_prefetch_b8)
            __builtin_prefetch(A  + (size_t)(m0) * lda + k + 32, 0, 3);
            __builtin_prefetch(Bt + (size_t)(n0) * ldb + k + 32, 0, 3);
        }
        v16bf a[4], b[4];
#pragma unroll
        for (int i = 0; i < 4; ++i) a[i] = load_mk_frag(A,  lda, m0 + 16 * i, k);
#pragma unroll
        for (int j = 0; j < 4; ++j) b[j] = load_mk_frag(Bt, ldb, n0 + 16 * j, k);
#pragma unroll
        for (int i = 0; i < 4; ++i)
#pragma unroll
            for (int j = 0; j < 4; ++j)
                r.c[i][j] = wmma_bf16(a[i], b[j], r.c[i][j]);
    }
    return r;
}

__device__ inline float gelu_f(float x) {
    const float k0 = 0.7978845608028654f, k1 = 0.044715f;
    return 0.5f * x * (1.f + tanhf(k0 * (x + k1 * x * x * x)));
}
__device__ inline float sigmoid_f(float x) { return 1.f / (1.f + expf(-x)); }

// ---------------------------------------------------------------------------
// Stage 0: weight conversion
// ---------------------------------------------------------------------------
__global__ void transpose_bf16_kernel(const float* __restrict__ in, bf16_t* __restrict__ out,
                                      int R, int Cc) {
    int idx = blockIdx.x * 256 + threadIdx.x;
    if (idx >= R * Cc) return;
    int r = idx / Cc, c = idx % Cc;
    out[(size_t)c * R + r] = (bf16_t)in[(size_t)r * Cc + c];
}
__global__ void convert_bf16_kernel(const float* __restrict__ in, bf16_t* __restrict__ out, int n) {
    int idx = blockIdx.x * 256 + threadIdx.x;
    if (idx < n) out[idx] = (bf16_t)in[idx];
}

// ---------------------------------------------------------------------------
// Stage 1: patchify (NCHW f32 -> bf16 patch matrices)
// patch feature f = (py*4+px)*64 + ch
// ---------------------------------------------------------------------------
__global__ void patchify_keys_kernel(const float* __restrict__ k0, const float* __restrict__ k1,
                                     bf16_t* __restrict__ Kp, bf16_t* __restrict__ KpT) {
    size_t idx = (size_t)blockIdx.x * 256 + threadIdx.x;
    if (idx >= (size_t)BB * NKT * DP) return;
    int f = (int)(idx % DP);
    size_t r = idx / DP;
    int n = (int)(r % NKT);
    int b = (int)(r / NKT);
    int s16 = f >> 6, ch = f & 63, py = s16 >> 2, px = s16 & 3;
    float v;
    if (n < NK0) {
        int hy = n / 32, wx = n % 32;
        v = k0[(((size_t)b * CCH + ch) * 128 + hy * 4 + py) * 128 + wx * 4 + px];
    } else {
        int n1 = n - NK0, hy = n1 / 24, wx = n1 % 24;
        v = k1[(((size_t)b * CCH + ch) * 96 + hy * 4 + py) * 96 + wx * 4 + px];
    }
    bf16_t bv = (bf16_t)v;
    Kp [((size_t)b * NKT + n) * DP + f]  = bv;
    KpT[((size_t)b * DP  + f) * NKT + n] = bv;
}

__global__ void patchify_queries_kernel(const float* __restrict__ q0, const float* __restrict__ q1,
                                        bf16_t* __restrict__ Qp) {
    size_t idx = (size_t)blockIdx.x * 256 + threadIdx.x;
    if (idx >= (size_t)BB * NQT * DP) return;
    int f = (int)(idx % DP);
    size_t r = idx / DP;
    int n = (int)(r % NQT);
    int b = (int)(r / NQT);
    int s16 = f >> 6, ch = f & 63, py = s16 >> 2, px = s16 & 3;
    float v;
    if (n < NQ0) {
        int hy = n / 32, wx = n % 32;
        v = q0[(((size_t)b * CCH + ch) * 128 + hy * 4 + py) * 128 + wx * 4 + px];
    } else {
        int n1 = n - NQ0, hy = n1 / 16, wx = n1 % 16;
        v = q1[(((size_t)b * CCH + ch) * 64 + hy * 4 + py) * 64 + wx * 4 + px];
    }
    Qp[idx] = (bf16_t)v;
}

// ---------------------------------------------------------------------------
// Stage 2: embedding GEMM: emb[b,h] = patch[b,:,h*128:+128] @ W + bias -> bf16
// grid: (rowTiles, B*NHEAD), block 256 (8 waves), wave tile = 32x32
// ---------------------------------------------------------------------------
__global__ void __launch_bounds__(256)
gemm_emb_kernel(const bf16_t* __restrict__ patch, const bf16_t* __restrict__ Wt,
                const float* __restrict__ bias, bf16_t* __restrict__ emb, int nRows) {
    const int wid = threadIdx.x >> 5, lane = threadIdx.x & 31;
    const int batch = blockIdx.y;
    const int b = batch >> 3, hd = batch & 7;
    const int t = blockIdx.x * 8 + wid;
    const int m0 = (t >> 3) * 32;        // ADIM/32 == 8 column tiles
    const int n0 = (t & 7) * 32;
    const bf16_t* A = patch + (size_t)b * nRows * DP + hd * DHEAD;
    Acc32 acc = wave_gemm32x32(A, DP, Wt, DHEAD, m0, n0, DHEAD);
    bf16_t* D = emb + (size_t)batch * nRows * ADIM;
    const int n = lane & 15, h = lane >> 4;
#pragma unroll
    for (int mi = 0; mi < 2; ++mi)
#pragma unroll
        for (int ni = 0; ni < 2; ++ni)
#pragma unroll
            for (int r = 0; r < 8; ++r) {
                int M = m0 + mi * 16 + h * 8 + r;
                int N = n0 + ni * 16 + n;
                D[(size_t)M * ADIM + N] = (bf16_t)(acc.c[mi][ni][r] + bias[N]);
            }
}

// ---------------------------------------------------------------------------
// Stage 3: S = Qemb @ Kemb^T  (f32 out; 1/sqrt(32) applied in softmax)
// 64x64 wave tiles: 20 x 25 = 500 tiles per (b,head)
// ---------------------------------------------------------------------------
__global__ void __launch_bounds__(256)
gemm_qk_kernel(const bf16_t* __restrict__ qemb, const bf16_t* __restrict__ kemb,
               float* __restrict__ S) {
    const int wid = threadIdx.x >> 5, lane = threadIdx.x & 31;
    const int batch = blockIdx.y;
    const int t = blockIdx.x * 8 + wid;
    if (t >= 20 * 25) return;
    const int m0 = (t / 25) * 64;
    const int n0 = (t % 25) * 64;
    const bf16_t* A  = qemb + (size_t)batch * NQT * ADIM;
    const bf16_t* Bt = kemb + (size_t)batch * NKT * ADIM;
    Acc64 acc = wave_gemm64x64(A, ADIM, Bt, ADIM, m0, n0, ADIM);
    float* D = S + (size_t)batch * NQT * NKT;
    const int n = lane & 15, h = lane >> 4;
#pragma unroll
    for (int mi = 0; mi < 4; ++mi)
#pragma unroll
        for (int ni = 0; ni < 4; ++ni)
#pragma unroll
            for (int r = 0; r < 8; ++r)
                D[(size_t)(m0 + mi * 16 + h * 8 + r) * NKT + n0 + ni * 16 + n]
                    = acc.c[mi][ni][r];
}

// ---------------------------------------------------------------------------
// Stage 4: row softmax over 1600 keys, P stored bf16
// ---------------------------------------------------------------------------
__global__ void __launch_bounds__(256)
softmax_kernel(const float* __restrict__ S, bf16_t* __restrict__ P) {
    __shared__ float buf[NKT];
    __shared__ float red[256];
    const size_t row = blockIdx.x;
    const float* s = S + row * NKT;
    bf16_t*      p = P + row * NKT;
    const float scale = 0.17677669529663687f;   // 1/sqrt(32)
    const int tid = threadIdx.x;
    float m = -3.4e38f;
    for (int i = tid; i < NKT; i += 256) { float z = s[i] * scale; buf[i] = z; m = fmaxf(m, z); }
    red[tid] = m; __syncthreads();
    for (int off = 128; off > 0; off >>= 1) {
        if (tid < off) red[tid] = fmaxf(red[tid], red[tid + off]);
        __syncthreads();
    }
    m = red[0]; __syncthreads();
    float sum = 0.f;
    for (int i = tid; i < NKT; i += 256) { float e = expf(buf[i] - m); buf[i] = e; sum += e; }
    red[tid] = sum; __syncthreads();
    for (int off = 128; off > 0; off >>= 1) {
        if (tid < off) red[tid] += red[tid + off];
        __syncthreads();
    }
    const float inv = 1.f / red[0];
    for (int i = tid; i < NKT; i += 256) p[i] = (bf16_t)(buf[i] * inv);
}

// ---------------------------------------------------------------------------
// Stage 5: wv = P[:, keys(src)] @ Kpatch(src); scatter f32 into NCHW image
// image channel = (head*2 + src)*64 + ch ; pixel from (query patch, py, px)
// 64x64 wave tiles: 20 x 16 = 320 tiles per (b,head)
// ---------------------------------------------------------------------------
__global__ void __launch_bounds__(256)
gemm_pv_kernel(const bf16_t* __restrict__ P, const bf16_t* __restrict__ KpT,
               float* __restrict__ V0, float* __restrict__ V1,
               int koff, int klen, int srcIdx) {
    const int wid = threadIdx.x >> 5, lane = threadIdx.x & 31;
    const int batch = blockIdx.y;
    const int b = batch >> 3, hd = batch & 7;
    const int t = blockIdx.x * 8 + wid;       // 20*16 = 320 tiles
    const int m0 = (t >> 4) * 64;
    const int n0 = (t & 15) * 64;
    const bf16_t* A  = P   + (size_t)batch * NQT * NKT + koff;
    const bf16_t* Bt = KpT + (size_t)b * DP * NKT + koff;
    Acc64 acc = wave_gemm64x64(A, NKT, Bt, NKT, m0, n0, klen);
    const int n = lane & 15, h = lane >> 4;
#pragma unroll
    for (int mi = 0; mi < 4; ++mi)
#pragma unroll
        for (int ni = 0; ni < 4; ++ni)
#pragma unroll
            for (int r = 0; r < 8; ++r) {
                const int M = m0 + mi * 16 + h * 8 + r;   // global query row
                const int c = n0 + ni * 16 + n;           // patch feature
                float* img; int nQ, wgrid, Himg;
                if (M < NQ0) { img = V0; nQ = M;       wgrid = 32; Himg = 128; }
                else         { img = V1; nQ = M - NQ0; wgrid = 16; Himg = 64;  }
                const int s16 = c >> 6, ch = c & 63, py = s16 >> 2, px = s16 & 3;
                const int hy = nQ / wgrid, wx = nQ % wgrid;
                const int co = (hd * 2 + srcIdx) * CCH + ch;
                img[(((size_t)b * 1024 + co) * Himg + hy * 4 + py) * Himg + wx * 4 + px]
                    = acc.c[mi][ni][r];
            }
}

// ---------------------------------------------------------------------------
// Stage 6: conv1 (grouped 1x1, 128 -> 32 per group) via WMMA + gelu
// A = bf16 weights [32 x 128] (group slice), B = NCHW activations [128 x HW]
// ---------------------------------------------------------------------------
__global__ void __launch_bounds__(256)
conv1_kernel(const float* __restrict__ V, const bf16_t* __restrict__ Wbf,
             const float* __restrict__ bias, float* __restrict__ Hout, int HW) {
    const int wid = threadIdx.x >> 5, lane = threadIdx.x & 31;
    const int batch = blockIdx.y;
    const int b = batch >> 3, g = batch & 7;
    const int n0 = (blockIdx.x * 8 + wid) * 32;
    const bf16_t* A   = Wbf + (size_t)g * 32 * 128;
    const float*  Bkn = V + ((size_t)b * 1024 + (size_t)g * 128) * HW;
    v8f acc[2][2];
#pragma unroll
    for (int i = 0; i < 2; ++i)
#pragma unroll
        for (int j = 0; j < 2; ++j)
            acc[i][j] = (v8f){0.f,0.f,0.f,0.f,0.f,0.f,0.f,0.f};
    for (int k = 0; k < 128; k += 32) {
        v16bf a0 = load_mk_frag(A, 128, 0,  k);
        v16bf a1 = load_mk_frag(A, 128, 16, k);
        v16bf b0 = load_kn_frag_f32(Bkn, HW, k, n0);
        v16bf b1 = load_kn_frag_f32(Bkn, HW, k, n0 + 16);
        acc[0][0] = wmma_bf16(a0, b0, acc[0][0]);
        acc[0][1] = wmma_bf16(a0, b1, acc[0][1]);
        acc[1][0] = wmma_bf16(a1, b0, acc[1][0]);
        acc[1][1] = wmma_bf16(a1, b1, acc[1][1]);
    }
    const int n = lane & 15, h = lane >> 4;
#pragma unroll
    for (int mi = 0; mi < 2; ++mi)
#pragma unroll
        for (int ni = 0; ni < 2; ++ni)
#pragma unroll
            for (int r = 0; r < 8; ++r) {
                const int o   = g * 32 + mi * 16 + h * 8 + r;
                const int pix = n0 + ni * 16 + n;
                Hout[((size_t)b * HID + o) * HW + pix] = gelu_f(acc[mi][ni][r] + bias[o]);
            }
}

// ---------------------------------------------------------------------------
// Stage 7: depthwise 3x3 (pad 1) + gelu
// ---------------------------------------------------------------------------
__global__ void dwconv_kernel(const float* __restrict__ Hin, const float* __restrict__ w,
                              const float* __restrict__ bias, float* __restrict__ Hout,
                              int Hs, int Ws) {
    size_t idx = (size_t)blockIdx.x * 256 + threadIdx.x;
    const size_t total = (size_t)BB * HID * Hs * Ws;
    if (idx >= total) return;
    const int x = (int)(idx % Ws);
    size_t r = idx / Ws;
    const int y = (int)(r % Hs); r /= Hs;
    const int o = (int)(r % HID);
    const int b = (int)(r / HID);
    const float* in = Hin + ((size_t)b * HID + o) * Hs * Ws;
    float a = bias[o];
#pragma unroll
    for (int ky = 0; ky < 3; ++ky) {
        const int yy = y + ky - 1;
        if (yy < 0 || yy >= Hs) continue;
#pragma unroll
        for (int kx = 0; kx < 3; ++kx) {
            const int xx = x + kx - 1;
            if (xx < 0 || xx >= Ws) continue;
            a += in[(size_t)yy * Ws + xx] * w[o * 9 + ky * 3 + kx];
        }
    }
    Hout[idx] = gelu_f(a);
}

// ---------------------------------------------------------------------------
// Stage 8: SE — spatial mean, MLP (silu, sigmoid)
// ---------------------------------------------------------------------------
__global__ void __launch_bounds__(256)
se_pool_kernel(const float* __restrict__ Hin, float* __restrict__ pool, int HW) {
    __shared__ float red[256];
    const int bc = blockIdx.x;                 // b*HID + ch
    const float* p = Hin + (size_t)bc * HW;
    float s = 0.f;
    for (int i = threadIdx.x; i < HW; i += 256) s += p[i];
    red[threadIdx.x] = s; __syncthreads();
    for (int off = 128; off > 0; off >>= 1) {
        if (threadIdx.x < off) red[threadIdx.x] += red[threadIdx.x + off];
        __syncthreads();
    }
    if (threadIdx.x == 0) pool[bc] = red[0] / (float)HW;
}

__global__ void __launch_bounds__(256)
se_mlp_kernel(const float* __restrict__ pool, const float* __restrict__ w1,
              const float* __restrict__ b1, const float* __restrict__ w2,
              const float* __restrict__ b2, float* __restrict__ scale) {
    __shared__ float mid[64];
    const int b = blockIdx.x;
    const int t = threadIdx.x;
    if (t < 64) {
        float a = b1[t];
        for (int i = 0; i < HID; ++i) a += pool[b * HID + i] * w1[t * HID + i];
        mid[t] = a * sigmoid_f(a);             // silu
    }
    __syncthreads();
    float a = b2[t];
    for (int j = 0; j < 64; ++j) a += mid[j] * w2[t * 64 + j];
    scale[b * HID + t] = sigmoid_f(a);
}

// ---------------------------------------------------------------------------
// Stage 9: conv2 (grouped 1x1, 32 -> 8 per group), scale applied, + bias
// ---------------------------------------------------------------------------
__global__ void conv2_kernel(const float* __restrict__ Hin, const float* __restrict__ scale,
                             const float* __restrict__ w, const float* __restrict__ bias,
                             float* __restrict__ Out, int HW) {
    size_t idx = (size_t)blockIdx.x * 256 + threadIdx.x;
    const size_t total = (size_t)BB * CCH * HW;
    if (idx >= total) return;
    const int pix = (int)(idx % HW);
    size_t r = idx / HW;
    const int co = (int)(r % CCH);
    const int b  = (int)(r / CCH);
    const int g  = co >> 3;
    float a = bias[co];
#pragma unroll 8
    for (int i = 0; i < 32; ++i) {
        const int ic = g * 32 + i;
        a += Hin[((size_t)b * HID + ic) * HW + pix] * scale[b * HID + ic] * w[co * 32 + i];
    }
    Out[idx] = a;
}

// ---------------------------------------------------------------------------
// Host-side orchestration
// ---------------------------------------------------------------------------
extern "C" void kernel_launch(void* const* d_in, const int* in_sizes, int n_in,
                              void* d_out, int out_size, void* d_ws, size_t ws_size,
                              hipStream_t stream) {
    (void)in_sizes; (void)n_in; (void)out_size; (void)ws_size;
    const float* key0    = (const float*)d_in[0];
    const float* key1    = (const float*)d_in[1];
    const float* query0  = (const float*)d_in[2];
    const float* query1  = (const float*)d_in[3];
    const float* Wk      = (const float*)d_in[4];
    const float* bk      = (const float*)d_in[5];
    const float* Wq      = (const float*)d_in[6];
    const float* bq      = (const float*)d_in[7];
    const float* conv1_w = (const float*)d_in[8];
    const float* conv1_b = (const float*)d_in[9];
    const float* dw_w    = (const float*)d_in[10];
    const float* dw_b    = (const float*)d_in[11];
    const float* se_w1   = (const float*)d_in[12];
    const float* se_b1   = (const float*)d_in[13];
    const float* se_w2   = (const float*)d_in[14];
    const float* se_b2   = (const float*)d_in[15];
    const float* conv2_w = (const float*)d_in[16];
    const float* conv2_b = (const float*)d_in[17];

    // ---- workspace carve-up (all chunks multiples of 256 bytes) ----
    char* ws = (char*)d_ws;
    size_t off = 0;
    auto alloc = [&](size_t bytes) -> void* {
        void* p = ws + off;
        off += (bytes + 255) & ~(size_t)255;
        return p;
    };
    bf16_t* Kpatch  = (bf16_t*)alloc((size_t)BB * NKT * DP * 2);     // 13.1 MB
    bf16_t* KpatchT = (bf16_t*)alloc((size_t)BB * DP * NKT * 2);     // 13.1 MB
    bf16_t* Qpatch  = (bf16_t*)alloc((size_t)BB * NQT * DP * 2);     // 10.5 MB
    bf16_t* WkT     = (bf16_t*)alloc((size_t)ADIM * DHEAD * 2);
    bf16_t* WqT     = (bf16_t*)alloc((size_t)ADIM * DHEAD * 2);
    bf16_t* W1bf    = (bf16_t*)alloc((size_t)HID * 128 * 2);
    bf16_t* Kemb    = (bf16_t*)alloc((size_t)BB * NHEAD * NKT * ADIM * 2); // 26 MB
    bf16_t* Qemb    = (bf16_t*)alloc((size_t)BB * NHEAD * NQT * ADIM * 2); // 21 MB
    float*  Sbuf    = (float*) alloc((size_t)BB * NHEAD * NQT * NKT * 4);  // 262 MB
    bf16_t* Pbuf    = (bf16_t*)alloc((size_t)BB * NHEAD * NQT * NKT * 2);  // 131 MB
    float*  V0      = (float*) alloc((size_t)BB * 1024 * 128 * 128 * 4);   // 268 MB
    float*  V1      = (float*) alloc((size_t)BB * 1024 * 64 * 64 * 4);     // 67 MB
    float*  H0a     = (float*) alloc((size_t)BB * HID * 16384 * 4);        // 67 MB
    float*  H0b     = (float*) alloc((size_t)BB * HID * 16384 * 4);
    float*  H1a     = (float*) alloc((size_t)BB * HID * 4096 * 4);         // 16.8 MB
    float*  H1b     = (float*) alloc((size_t)BB * HID * 4096 * 4);
    float*  pool0   = (float*) alloc((size_t)BB * HID * 4);
    float*  scl0    = (float*) alloc((size_t)BB * HID * 4);
    float*  pool1   = (float*) alloc((size_t)BB * HID * 4);
    float*  scl1    = (float*) alloc((size_t)BB * HID * 4);

    float* out0 = (float*)d_out;                                // (4,64,128,128)
    float* out1 = (float*)d_out + (size_t)BB * CCH * 128 * 128; // (4,64,64,64)

    // ---- 0: weight prep ----
    transpose_bf16_kernel<<<(DHEAD * ADIM + 255) / 256, 256, 0, stream>>>(Wk, WkT, DHEAD, ADIM);
    transpose_bf16_kernel<<<(DHEAD * ADIM + 255) / 256, 256, 0, stream>>>(Wq, WqT, DHEAD, ADIM);
    convert_bf16_kernel<<<(HID * 128 + 255) / 256, 256, 0, stream>>>(conv1_w, W1bf, HID * 128);

    // ---- 1: patchify ----
    patchify_keys_kernel<<<(unsigned)(((size_t)BB * NKT * DP + 255) / 256), 256, 0, stream>>>(
        key0, key1, Kpatch, KpatchT);
    patchify_queries_kernel<<<(unsigned)(((size_t)BB * NQT * DP + 255) / 256), 256, 0, stream>>>(
        query0, query1, Qpatch);

    // ---- 2: embeddings (WMMA, 32x32 tiles, K=128) ----
    gemm_emb_kernel<<<dim3(NKT / 32, BB * NHEAD), 256, 0, stream>>>(Kpatch, WkT, bk, Kemb, NKT);
    gemm_emb_kernel<<<dim3(NQT / 32, BB * NHEAD), 256, 0, stream>>>(Qpatch, WqT, bq, Qemb, NQT);

    // ---- 3: S = Q K^T (WMMA, 64x64 tiles; 500 tiles -> 63 blocks of 8 waves) ----
    gemm_qk_kernel<<<dim3(63, BB * NHEAD), 256, 0, stream>>>(Qemb, Kemb, Sbuf);

    // ---- 4: softmax ----
    softmax_kernel<<<BB * NHEAD * NQT, 256, 0, stream>>>(Sbuf, Pbuf);

    // ---- 5: P @ V, scattered into MBConv NCHW input (WMMA, 64x64 tiles) ----
    gemm_pv_kernel<<<dim3(40, BB * NHEAD), 256, 0, stream>>>(Pbuf, KpatchT, V0, V1, 0,   NK0, 0);
    gemm_pv_kernel<<<dim3(40, BB * NHEAD), 256, 0, stream>>>(Pbuf, KpatchT, V0, V1, NK0, NK1, 1);

    // ---- 6: conv1 grouped 1x1 + gelu (WMMA) ----
    conv1_kernel<<<dim3(64, BB * NHEAD), 256, 0, stream>>>(V0, W1bf, conv1_b, H0a, 16384);
    conv1_kernel<<<dim3(16, BB * NHEAD), 256, 0, stream>>>(V1, W1bf, conv1_b, H1a, 4096);

    // ---- 7: depthwise 3x3 + gelu ----
    dwconv_kernel<<<(unsigned)(((size_t)BB * HID * 16384 + 255) / 256), 256, 0, stream>>>(
        H0a, dw_w, dw_b, H0b, 128, 128);
    dwconv_kernel<<<(unsigned)(((size_t)BB * HID * 4096 + 255) / 256), 256, 0, stream>>>(
        H1a, dw_w, dw_b, H1b, 64, 64);

    // ---- 8: SE ----
    se_pool_kernel<<<BB * HID, 256, 0, stream>>>(H0b, pool0, 16384);
    se_pool_kernel<<<BB * HID, 256, 0, stream>>>(H1b, pool1, 4096);
    se_mlp_kernel<<<BB, 256, 0, stream>>>(pool0, se_w1, se_b1, se_w2, se_b2, scl0);
    se_mlp_kernel<<<BB, 256, 0, stream>>>(pool1, se_w1, se_b1, se_w2, se_b2, scl1);

    // ---- 9: conv2 grouped 1x1 -> outputs ----
    conv2_kernel<<<(unsigned)(((size_t)BB * CCH * 16384 + 255) / 256), 256, 0, stream>>>(
        H0b, scl0, conv2_w, conv2_b, out0, 16384);
    conv2_kernel<<<(unsigned)(((size_t)BB * CCH * 4096 + 255) / 256), 256, 0, stream>>>(
        H1b, scl1, conv2_w, conv2_b, out1, 4096);
}
